// HybridNodeBlock_48034914239039
// MI455X (gfx1250) — matrix operand order repaired and checked by reading, not compiled, not run on hardware
//
#include <hip/hip_runtime.h>
#include <hip/hip_bf16.h>
#include <stdint.h>

#define DD   128
#define MT   128          // rows per block in fused MLP kernel
#define KP1  392          // padded pitch (elems) for A / W1t  (384 + 8)
#define KP2  136          // padded pitch (elems) for h / W2t / W3t (128 + 8)

typedef __attribute__((ext_vector_type(16))) __bf16 v16bf;
typedef __attribute__((ext_vector_type(8)))  float  v8f;

union Frag { v16bf v; uint4 u[2]; };

#define WMMA_BF16(A, B, C) __builtin_amdgcn_wmma_f32_16x16x32_bf16( \
        false, (A), false, (B), (short)0, (C), false, false)

#if __has_builtin(__builtin_amdgcn_sched_barrier)
#define SCHED_FENCE() __builtin_amdgcn_sched_barrier(0)
#else
#define SCHED_FENCE() asm volatile("" ::: "memory")
#endif

__device__ __forceinline__ uint16_t f2bf(float f) {
    uint32_t u = __builtin_bit_cast(uint32_t, f);
    u += 0x7FFFu + ((u >> 16) & 1u);          // round-to-nearest-even
    return (uint16_t)(u >> 16);
}
__device__ __forceinline__ uint32_t pack2(float a, float b) {
    return (uint32_t)f2bf(a) | ((uint32_t)f2bf(b) << 16);
}

// ---------------- weight cast + transpose: dst[n*Kp + k] = bf16(src[k*128 + n])
__global__ void wprep_kernel(const float* __restrict__ src, uint16_t* __restrict__ dst,
                             int K, int Kp) {
    int i = blockIdx.x * blockDim.x + threadIdx.x;
    if (i >= DD * K) return;
    int n = i / K;
    int k = i - n * K;
    dst[n * Kp + k] = f2bf(src[(size_t)k * DD + n]);
}

// ---------------- zero fill (float4 granularity)
__global__ void zero_kernel(float4* __restrict__ p, long long n4) {
    long long i = (long long)blockIdx.x * blockDim.x + threadIdx.x;
    if (i < n4) p[i] = make_float4(0.f, 0.f, 0.f, 0.f);
}

// ---------------- scatter-sum: one wave per edge, 4 floats per lane
__global__ void scatter_kernel(const float* __restrict__ eattr,
                               const int* __restrict__ eidx,
                               int nE, float* __restrict__ agg) {
    int e = blockIdx.x * 8 + (threadIdx.x >> 5);
    if (e >= nE) return;
    int lane = threadIdx.x & 31;
    int recv = eidx[nE + e];                              // edge_index[1][e]
    float4 v = ((const float4*)(eattr + (size_t)e * DD))[lane];
    float* dst = agg + (size_t)recv * DD + lane * 4;
    atomicAdd(dst + 0, v.x);
    atomicAdd(dst + 1, v.y);
    atomicAdd(dst + 2, v.z);
    atomicAdd(dst + 3, v.w);
}

// ---------------- fused MLP + LayerNorm, WMMA bf16 -> f32
__global__ __launch_bounds__(256) void mlp_fused_kernel(
    const float* __restrict__ x,
    const float* __restrict__ magg,
    const float* __restrict__ wagg,
    const uint16_t* __restrict__ w1t,   // [128][KP1] bf16 (k-contiguous)
    const uint16_t* __restrict__ w2t,   // [128][KP2]
    const uint16_t* __restrict__ w3t,   // [128][KP2]
    const float* __restrict__ b1,
    const float* __restrict__ b2,
    const float* __restrict__ b3,
    const float* __restrict__ gamma,
    const float* __restrict__ beta,
    float* __restrict__ out,
    int nnodes)
{
    extern __shared__ char smem[];
    uint16_t* sA  = (uint16_t*)(smem);                                  // [MT][KP1], reused as h2 [MT][KP2]
    uint16_t* sW1 = (uint16_t*)(smem + (size_t)MT * KP1 * 2);           // [128][KP1]
    uint16_t* sW2 = sW1 + (size_t)DD * KP1;                             // [128][KP2]
    uint16_t* sW3 = sW2 + (size_t)DD * KP2;                             // [128][KP2]
    uint16_t* sH  = sW3 + (size_t)DD * KP2;                             // [MT][KP2]

    const int tid  = threadIdx.x;
    const int wave = tid >> 5;
    const int lane = tid & 31;
    const int half = lane >> 4;       // 0: lanes 0-15, 1: lanes 16-31
    const int ln16 = lane & 15;
    const int r0   = blockIdx.x * MT;

    // ---- stage bf16 weights (L2-resident) into LDS, uint4 copies
    {
        const uint4* s1 = (const uint4*)w1t; uint4* d1 = (uint4*)sW1;
        const int n1 = (DD * KP1 * 2) / 16;
        for (int i = tid; i < n1; i += 256) d1[i] = s1[i];
        const uint4* s2 = (const uint4*)w2t; uint4* d2 = (uint4*)sW2;
        const uint4* s3 = (const uint4*)w3t; uint4* d3 = (uint4*)sW3;
        const int n2 = (DD * KP2 * 2) / 16;
        for (int i = tid; i < n2; i += 256) d2[i] = s2[i];
        for (int i = tid; i < n2; i += 256) d3[i] = s3[i];
    }

    // ---- gather [x | mesh_agg | world_agg] rows -> bf16 A-tile in LDS
    for (int i = tid; i < MT * 96; i += 256) {        // 96 float4 per row (3 segs x 32)
        int row = i / 96;
        int q   = i - row * 96;
        int seg = q >> 5;
        int c4  = q & 31;
        int node = r0 + row;
        float4 v = make_float4(0.f, 0.f, 0.f, 0.f);
        if (node < nnodes) {
            const float* src = (seg == 0) ? x : (seg == 1) ? magg : wagg;
            v = *(const float4*)(src + (size_t)node * DD + c4 * 4);
        }
        *(uint2*)(sA + (size_t)row * KP1 + seg * DD + c4 * 4) =
            make_uint2(pack2(v.x, v.y), pack2(v.z, v.w));
    }

    // ---- per-lane column constants
    float bb1[8], bb2[8], bb3[8], gg[8], be[8];
    #pragma unroll
    for (int t = 0; t < 8; ++t) {
        int c = ln16 + 16 * t;
        bb1[t] = b1[c]; bb2[t] = b2[c]; bb3[t] = b3[c];
        gg[t] = gamma[c]; be[t] = beta[c];
    }

    __syncthreads();

    const int rowb = wave * 16 + ln16;   // A-fragment row for this lane
    v8f acc[8] = {};
    v8f zeroV = {};

    // ================= GEMM1: [16x384] @ [384x128] =================
    #pragma unroll
    for (int k0 = 0; k0 < 3 * DD; k0 += 32) {
        Frag a;
        const uint16_t* ap = sA + (size_t)rowb * KP1 + k0 + half * 16;
        a.u[0] = *(const uint4*)(ap);
        a.u[1] = *(const uint4*)(ap + 8);
        Frag bf[8];                       // batch all LDS loads of this k-step
        #pragma unroll
        for (int t = 0; t < 8; ++t) {
            const uint16_t* bp = sW1 + (size_t)(ln16 + 16 * t) * KP1 + k0 + half * 16;
            bf[t].u[0] = *(const uint4*)(bp);
            bf[t].u[1] = *(const uint4*)(bp + 8);
        }
        SCHED_FENCE();                    // all loads issued before any WMMA -> 8 live frags
        #pragma unroll
        for (int t = 0; t < 8; ++t)
            acc[t] = WMMA_BF16(a.v, bf[t].v, acc[t]);
        SCHED_FENCE();
    }
    // bias1 + ReLU -> h1 (bf16) in sH
    #pragma unroll
    for (int t = 0; t < 8; ++t) {
        #pragma unroll
        for (int j = 0; j < 8; ++j) {
            float v = acc[t][j] + bb1[t];
            v = v > 0.f ? v : 0.f;
            int row = wave * 16 + j + half * 8;
            sH[(size_t)row * KP2 + ln16 + 16 * t] = f2bf(v);
        }
    }
    __syncthreads();

    // ================= GEMM2: [16x128] @ [128x128] =================
    #pragma unroll
    for (int t = 0; t < 8; ++t) acc[t] = zeroV;
    #pragma unroll
    for (int k0 = 0; k0 < DD; k0 += 32) {
        Frag a;
        const uint16_t* ap = sH + (size_t)rowb * KP2 + k0 + half * 16;
        a.u[0] = *(const uint4*)(ap);
        a.u[1] = *(const uint4*)(ap + 8);
        Frag bf[8];
        #pragma unroll
        for (int t = 0; t < 8; ++t) {
            const uint16_t* bp = sW2 + (size_t)(ln16 + 16 * t) * KP2 + k0 + half * 16;
            bf[t].u[0] = *(const uint4*)(bp);
            bf[t].u[1] = *(const uint4*)(bp + 8);
        }
        SCHED_FENCE();
        #pragma unroll
        for (int t = 0; t < 8; ++t)
            acc[t] = WMMA_BF16(a.v, bf[t].v, acc[t]);
        SCHED_FENCE();
    }
    // bias2 + ReLU -> h2 (bf16) reusing sA
    uint16_t* sH2 = sA;
    #pragma unroll
    for (int t = 0; t < 8; ++t) {
        #pragma unroll
        for (int j = 0; j < 8; ++j) {
            float v = acc[t][j] + bb2[t];
            v = v > 0.f ? v : 0.f;
            int row = wave * 16 + j + half * 8;
            sH2[(size_t)row * KP2 + ln16 + 16 * t] = f2bf(v);
        }
    }
    __syncthreads();

    // ================= GEMM3: [16x128] @ [128x128] =================
    #pragma unroll
    for (int t = 0; t < 8; ++t) acc[t] = zeroV;
    #pragma unroll
    for (int k0 = 0; k0 < DD; k0 += 32) {
        Frag a;
        const uint16_t* ap = sH2 + (size_t)rowb * KP2 + k0 + half * 16;
        a.u[0] = *(const uint4*)(ap);
        a.u[1] = *(const uint4*)(ap + 8);
        Frag bf[8];
        #pragma unroll
        for (int t = 0; t < 8; ++t) {
            const uint16_t* bp = sW3 + (size_t)(ln16 + 16 * t) * KP2 + k0 + half * 16;
            bf[t].u[0] = *(const uint4*)(bp);
            bf[t].u[1] = *(const uint4*)(bp + 8);
        }
        SCHED_FENCE();
        #pragma unroll
        for (int t = 0; t < 8; ++t)
            acc[t] = WMMA_BF16(a.v, bf[t].v, acc[t]);
        SCHED_FENCE();
    }

    // ---- bias3 + in-register LayerNorm over the 128 columns
    #pragma unroll
    for (int t = 0; t < 8; ++t) {
        #pragma unroll
        for (int j = 0; j < 8; ++j) acc[t][j] += bb3[t];
    }
    float p[8], q[8];
    #pragma unroll
    for (int j = 0; j < 8; ++j) {
        float s = 0.f, s2 = 0.f;
        #pragma unroll
        for (int t = 0; t < 8; ++t) { float v = acc[t][j]; s += v; s2 += v * v; }
        p[j] = s; q[j] = s2;
    }
    #pragma unroll
    for (int m = 1; m <= 8; m <<= 1) {     // reduce within each 16-lane half
        #pragma unroll
        for (int j = 0; j < 8; ++j) {
            p[j] += __shfl_xor(p[j], m, 32);
            q[j] += __shfl_xor(q[j], m, 32);
        }
    }
    #pragma unroll
    for (int j = 0; j < 8; ++j) {
        float mean = p[j] * (1.0f / 128.0f);
        float var  = q[j] * (1.0f / 128.0f) - mean * mean;
        float rstd = rsqrtf(var + 1e-5f);
        int grow = r0 + wave * 16 + j + half * 8;
        if (grow < nnodes) {
            float* op = out + (size_t)grow * DD;
            #pragma unroll
            for (int t = 0; t < 8; ++t)
                op[ln16 + 16 * t] = (acc[t][j] - mean) * rstd * gg[t] + be[t];
        }
    }
}

extern "C" void kernel_launch(void* const* d_in, const int* in_sizes, int n_in,
                              void* d_out, int out_size, void* d_ws, size_t ws_size,
                              hipStream_t stream) {
    const float* x     = (const float*)d_in[0];
    const float* eattr = (const float*)d_in[1];
    const int*   eidx  = (const int*)d_in[2];
    const float* wattr = (const float*)d_in[3];
    const int*   widx  = (const int*)d_in[4];
    const float* W1    = (const float*)d_in[5];
    const float* b1    = (const float*)d_in[6];
    const float* W2    = (const float*)d_in[7];
    const float* b2    = (const float*)d_in[8];
    const float* W3    = (const float*)d_in[9];
    const float* b3    = (const float*)d_in[10];
    const float* gam   = (const float*)d_in[11];
    const float* bet   = (const float*)d_in[12];

    const int nnodes = in_sizes[0] / DD;
    const int nMesh  = in_sizes[1] / DD;
    const int nWorld = in_sizes[3] / DD;

    char* ws = (char*)d_ws;
    uint16_t* w1t = (uint16_t*)ws;                                      // 128*KP1 bf16
    uint16_t* w2t = (uint16_t*)(ws + (size_t)DD * KP1 * 2);             // 128*KP2
    uint16_t* w3t = (uint16_t*)(ws + (size_t)DD * KP1 * 2 + (size_t)DD * KP2 * 2);
    float* magg = (float*)(ws + (size_t)DD * KP1 * 2 + 2 * (size_t)DD * KP2 * 2);
    float* waggp = magg + (size_t)nnodes * DD;

    // 1) one-time weight cast/transpose into workspace (bf16, k-contiguous, padded)
    wprep_kernel<<<(DD * 3 * DD + 255) / 256, 256, 0, stream>>>(W1, w1t, 3 * DD, KP1);
    wprep_kernel<<<(DD * DD + 255) / 256, 256, 0, stream>>>(W2, w2t, DD, KP2);
    wprep_kernel<<<(DD * DD + 255) / 256, 256, 0, stream>>>(W3, w3t, DD, KP2);

    // 2) zero aggregation buffers
    long long n4 = (long long)nnodes * DD * 2 / 4;
    zero_kernel<<<(unsigned)((n4 + 255) / 256), 256, 0, stream>>>((float4*)magg, n4);

    // 3) scatter-sum both edge sets (atomics resolve in 192MB L2)
    scatter_kernel<<<(nMesh + 7) / 8, 256, 0, stream>>>(eattr, eidx, nMesh, magg);
    scatter_kernel<<<(nWorld + 7) / 8, 256, 0, stream>>>(wattr, widx, nWorld, waggp);

    // 4) fused concat + 3-layer MLP + LayerNorm
    int nblk = (nnodes + MT - 1) / MT;
    size_t smem = (size_t)MT * KP1 * 2      // A tile (reused as h2)
                + (size_t)DD * KP1 * 2      // W1t
                + (size_t)DD * KP2 * 2 * 2  // W2t + W3t
                + (size_t)MT * KP2 * 2;     // h1
    mlp_fused_kernel<<<nblk, 256, smem, stream>>>(x, magg, waggp, w1t, w2t, w3t,
                                                  b1, b2, b3, gam, bet,
                                                  (float*)d_out, nnodes);
}
